// GroupedQueryAttention_51874615001432
// MI455X (gfx1250) — matrix operand order repaired and checked by reading, not compiled, not run on hardware
//
#include <hip/hip_runtime.h>
#include <stddef.h>

typedef unsigned int   u32;
typedef unsigned short u16;

typedef __bf16 v16bf __attribute__((ext_vector_type(16)));
typedef float  v8f   __attribute__((ext_vector_type(8)));
typedef u32    u32x8 __attribute__((ext_vector_type(8)));
typedef int    v4i   __attribute__((ext_vector_type(4)));

#if defined(__has_builtin)
#if __has_builtin(__builtin_amdgcn_global_load_async_to_lds_b128) && \
    __has_builtin(__builtin_amdgcn_s_wait_asynccnt)
#define HAVE_ASYNC_LDS 1
#endif
#endif

#ifdef HAVE_ASYNC_LDS
// global -> LDS async copy of 16 bytes (GLOBAL_LOAD_ASYNC_TO_LDS_B128, ASYNCcnt).
// Builtin takes (global v4i* src, lds v4i* dst, imm offset, imm cpol).
#define ASYNC_CP16(dstLds, srcGlob)                                        \
    __builtin_amdgcn_global_load_async_to_lds_b128(                        \
        (__attribute__((address_space(1))) v4i*)(size_t)(srcGlob),         \
        (__attribute__((address_space(3))) v4i*)(size_t)(dstLds), 0, 0)
#endif

// ---------- helpers ----------
static __device__ __forceinline__ u16 f32_to_bf16(float f) {
    __bf16 h = (__bf16)f;                 // native cvt (RNE) on gfx1250
    return __builtin_bit_cast(u16, h);
}

// ---------- kernel 0: fp32 -> bf16 elementwise ----------
__global__ void cvt_f32_bf16(const float* __restrict__ in, u16* __restrict__ out, int n) {
    for (int i = blockIdx.x * blockDim.x + threadIdx.x; i < n; i += gridDim.x * blockDim.x)
        out[i] = f32_to_bf16(in[i]);
}

// ---------- kernel 1: tiled WMMA GEMM  C[M,N] = A[M,K] * W[N,K]^T ----------
// mode 0: bf16 row-major out; mode 1: f32 row-major out;
// mode 2: bf16 out stored transposed per kv-head: Vt[b][h][d][s]  (N must be Hkv*64)
#define BM 64
#define BN 128
#define BK 32

__global__ __launch_bounds__(256) void gemm_wmma_bf16(
    const float* __restrict__ Af,      // non-null -> A is fp32 (converted on load)
    const u16*   __restrict__ Ab,      // else A is bf16
    const u16*   __restrict__ Wb,      // weights bf16, [N, K] row-major ( == B^T )
    void*        __restrict__ Cout,
    int M, int N, int K, int mode, int Sdim)
{
    __shared__ u16 As[BM][BK + 2];
    __shared__ u16 Bs[BN][BK + 2];

    const int tid  = threadIdx.x;
    const int lane = tid & 31;
    const int wid  = tid >> 5;          // 0..7
    const int wm   = wid & 1;           // 2 M-slabs of 32
    const int wn   = wid >> 1;          // 4 N-slabs of 32
    const int m0   = blockIdx.y * BM;
    const int n0   = blockIdx.x * BN;

    v8f acc[2][2] = {};

    for (int k0 = 0; k0 < K; k0 += BK) {
        if (k0 + BK < K) {
            if (Af) __builtin_prefetch(Af + (size_t)(m0 + (tid & 63)) * K + k0 + BK, 0, 1);
            else    __builtin_prefetch(Ab + (size_t)(m0 + (tid & 63)) * K + k0 + BK, 0, 1);
            __builtin_prefetch(Wb + (size_t)(n0 + (tid & 127)) * K + k0 + BK, 0, 1);
        }
        // --- stage A tile (64x32), 8 elems / thread ---
        {
            int e = tid * 8;
            int r = e >> 5, c = e & 31;
            if (Af) {
                const float* ap = Af + (size_t)(m0 + r) * K + k0 + c;
                u32* dst = (u32*)&As[r][c];
                #pragma unroll
                for (int i = 0; i < 4; i++)
                    dst[i] = (u32)f32_to_bf16(ap[2 * i]) |
                             ((u32)f32_to_bf16(ap[2 * i + 1]) << 16);
            } else {
#ifdef HAVE_ASYNC_LDS
                ASYNC_CP16(&As[r][c], Ab + (size_t)(m0 + r) * K + k0 + c);
#else
                const uint4 q4 = *(const uint4*)(Ab + (size_t)(m0 + r) * K + k0 + c);
                u32* dst = (u32*)&As[r][c];
                dst[0] = q4.x; dst[1] = q4.y; dst[2] = q4.z; dst[3] = q4.w;
#endif
            }
        }
        // --- stage B tile (128x32), 16 elems / thread ---
        #pragma unroll
        for (int cc = 0; cc < 2; cc++) {
            int e = (tid + cc * 256) * 8;
            int r = e >> 5, c = e & 31;
#ifdef HAVE_ASYNC_LDS
            ASYNC_CP16(&Bs[r][c], Wb + (size_t)(n0 + r) * K + k0 + c);
#else
            const uint4 q4 = *(const uint4*)(Wb + (size_t)(n0 + r) * K + k0 + c);
            u32* dst = (u32*)&Bs[r][c];
            dst[0] = q4.x; dst[1] = q4.y; dst[2] = q4.z; dst[3] = q4.w;
#endif
        }
#ifdef HAVE_ASYNC_LDS
        __builtin_amdgcn_s_wait_asynccnt(0);   // drain async LDS writes before barrier
#endif
        __syncthreads();

        // --- fragments (ISA VGPR layouts) ---
        v16bf afr[2], bfr[2];
        #pragma unroll
        for (int i = 0; i < 2; i++) {
            int row = wm * 32 + i * 16 + (lane & 15);
            int kg  = (lane >> 4) * 8;
            u32x8 t;
            #pragma unroll
            for (int v = 0; v < 8; v++) {
                int k = ((v & 4) ? 16 : 0) + kg + (v & 3) * 2;
                t[v] = *(const u32*)&As[row][k];
            }
            afr[i] = __builtin_bit_cast(v16bf, t);
        }
        #pragma unroll
        for (int j = 0; j < 2; j++) {
            int col = wn * 32 + j * 16 + (lane & 15);
            int kb  = (lane >> 4) * 8;
            u32x8 t;
            #pragma unroll
            for (int v = 0; v < 8; v++)
                t[v] = *(const u32*)&Bs[col][kb + ((v & 4) ? 16 : 0) + (v & 3) * 2];
            bfr[j] = __builtin_bit_cast(v16bf, t);
        }
        #pragma unroll
        for (int i = 0; i < 2; i++)
            #pragma unroll
            for (int j = 0; j < 2; j++)
                acc[i][j] = __builtin_amdgcn_wmma_f32_16x16x32_bf16(
                    false, afr[i], false, bfr[j], (short)0, acc[i][j], false, false);
        __syncthreads();
    }

    // --- store (WMMA C layout: M = r + 8*(lane>=16), N = lane&15) ---
    #pragma unroll
    for (int i = 0; i < 2; i++)
        #pragma unroll
        for (int j = 0; j < 2; j++)
            #pragma unroll
            for (int r = 0; r < 8; r++) {
                int m = m0 + wm * 32 + i * 16 + r + ((lane >> 4) ? 8 : 0);
                int n = n0 + wn * 32 + j * 16 + (lane & 15);
                float val = acc[i][j][r];
                if (mode == 1) {
                    ((float*)Cout)[(size_t)m * N + n] = val;
                } else if (mode == 0) {
                    ((u16*)Cout)[(size_t)m * N + n] = f32_to_bf16(val);
                } else { // mode 2: transposed V store
                    int b = m / Sdim, s = m % Sdim;
                    int h = n >> 6,  d = n & 63;
                    ((u16*)Cout)[(((size_t)b * (N >> 6) + h) * 64 + d) * (size_t)Sdim + s] =
                        f32_to_bf16(val);
                }
            }
}

// ---------- kernel 2: flash attention (GQA), all-WMMA ----------
// Qb: [B*S, 2048] bf16; Kb: [B*S, 512] bf16; Vt: [B][8][64][S] bf16; Ob: [B*S, 2048] bf16
__global__ __launch_bounds__(128) void gqa_flash_attn(
    const u16* __restrict__ Qb, const u16* __restrict__ Kb,
    const u16* __restrict__ Vt, u16* __restrict__ Ob)
{
    constexpr int S = 2048;
    __shared__ u16 Plds[4][16][68];     // per-wave P tile, padded

    const int tid  = threadIdx.x;
    const int lane = tid & 31;
    const int wid  = tid >> 5;          // 0..3 (q-row slab)
    const int head = blockIdx.y & 31;
    const int b    = blockIdx.y >> 5;
    const int kvh  = head >> 2;         // grouped KV head
    const int q0   = blockIdx.x * 64 + wid * 16;

    const float sfac = 0.125f * 1.44269504088896f; // 1/sqrt(64) * log2(e)

    // Q fragments (A layout), 16 rows x 64 k
    v16bf qf[2];
    {
        int row = q0 + (lane & 15);
        int kg  = (lane >> 4) * 8;
        const u16* qp = Qb + ((size_t)b * S + row) * 2048 + head * 64;
        #pragma unroll
        for (int hf = 0; hf < 2; hf++) {
            u32x8 t;
            #pragma unroll
            for (int v = 0; v < 8; v++) {
                int d = hf * 32 + ((v & 4) ? 16 : 0) + kg + (v & 3) * 2;
                t[v] = *(const u32*)(qp + d);
            }
            qf[hf] = __builtin_bit_cast(v16bf, t);
        }
    }

    v8f   o[4] = {};
    float mrow[8], lrow[8];
    #pragma unroll
    for (int r = 0; r < 8; r++) { mrow[r] = -1e30f; lrow[r] = 0.f; }

    for (int kblk = 0; kblk < S; kblk += 64) {
        // ---- scores: S = Q * K^T (16 x 64) ----
        v8f sc[4] = {};
        #pragma unroll
        for (int j = 0; j < 4; j++) {
            int keypos = kblk + j * 16 + (lane & 15);
            const u16* kp = Kb + ((size_t)b * S + keypos) * 512 + kvh * 64;
            int kb = (lane >> 4) * 8;
            #pragma unroll
            for (int hf = 0; hf < 2; hf++) {
                u32x8 t;
                #pragma unroll
                for (int v = 0; v < 8; v++)
                    t[v] = *(const u32*)(kp + hf * 32 + kb + ((v & 4) ? 16 : 0) + (v & 3) * 2);
                v16bf kf = __builtin_bit_cast(v16bf, t);
                sc[j] = __builtin_amdgcn_wmma_f32_16x16x32_bf16(
                    false, qf[hf], false, kf, (short)0, sc[j], false, false);
            }
        }
        // ---- online softmax (per-row stats over 16 lanes of each half) ----
        float corr[8];
        #pragma unroll
        for (int r = 0; r < 8; r++) {
            float v = fmaxf(fmaxf(sc[0][r], sc[1][r]), fmaxf(sc[2][r], sc[3][r]));
            v *= sfac;
            #pragma unroll
            for (int off = 1; off < 16; off <<= 1)
                v = fmaxf(v, __shfl_xor(v, off, 32));
            float nm = fmaxf(mrow[r], v);
            corr[r]  = exp2f(mrow[r] - nm);
            mrow[r]  = nm;
        }
        #pragma unroll
        for (int r = 0; r < 8; r++) {
            float rs = 0.f;
            int prow = r + ((lane >> 4) ? 8 : 0);
            #pragma unroll
            for (int j = 0; j < 4; j++) {
                float p = exp2f(sc[j][r] * sfac - mrow[r]);
                rs += p;
                Plds[wid][prow][j * 16 + (lane & 15)] = f32_to_bf16(p);
            }
            #pragma unroll
            for (int off = 1; off < 16; off <<= 1)
                rs += __shfl_xor(rs, off, 32);
            lrow[r] = lrow[r] * corr[r] + rs;
            #pragma unroll
            for (int j = 0; j < 4; j++) o[j][r] *= corr[r];
        }
        // ---- P fragments from LDS (C-layout -> A-layout; wave-private, LDS in-order) ----
        v16bf pf[2];
        {
            int row = lane & 15;
            int kg  = (lane >> 4) * 8;
            #pragma unroll
            for (int hf = 0; hf < 2; hf++) {
                u32x8 t;
                #pragma unroll
                for (int v = 0; v < 8; v++) {
                    int kv = hf * 32 + ((v & 4) ? 16 : 0) + kg + (v & 3) * 2;
                    t[v] = *(const u32*)&Plds[wid][row][kv];
                }
                pf[hf] = __builtin_bit_cast(v16bf, t);
            }
        }
        // ---- O += P * V  (V pre-transposed: contiguous k pairs) ----
        #pragma unroll
        for (int j = 0; j < 4; j++) {
            int n = j * 16 + (lane & 15);
            const u16* vp = Vt + (((size_t)b * 8 + kvh) * 64 + n) * (size_t)S + kblk;
            int kb = (lane >> 4) * 8;
            #pragma unroll
            for (int hf = 0; hf < 2; hf++) {
                u32x8 t;
                #pragma unroll
                for (int v = 0; v < 8; v++)
                    t[v] = *(const u32*)(vp + hf * 32 + kb + ((v & 4) ? 16 : 0) + (v & 3) * 2);
                v16bf vf = __builtin_bit_cast(v16bf, t);
                o[j] = __builtin_amdgcn_wmma_f32_16x16x32_bf16(
                    false, pf[hf], false, vf, (short)0, o[j], false, false);
            }
        }
    }
    // ---- epilogue: normalize and store bf16 ----
    #pragma unroll
    for (int j = 0; j < 4; j++)
        #pragma unroll
        for (int r = 0; r < 8; r++) {
            int m = q0 + r + ((lane >> 4) ? 8 : 0);
            int d = j * 16 + (lane & 15);
            Ob[((size_t)b * S + m) * 2048 + head * 64 + d] = f32_to_bf16(o[j][r] / lrow[r]);
        }
}

// ---------- host ----------
extern "C" void kernel_launch(void* const* d_in, const int* in_sizes, int n_in,
                              void* d_out, int out_size, void* d_ws, size_t ws_size,
                              hipStream_t stream) {
    (void)in_sizes; (void)n_in; (void)out_size; (void)ws_size;
    const float* query = (const float*)d_in[0];
    const float* key   = (const float*)d_in[1];
    const float* value = (const float*)d_in[2];
    const float* W_q   = (const float*)d_in[3];
    const float* W_k   = (const float*)d_in[4];
    const float* W_v   = (const float*)d_in[5];
    const float* W_o   = (const float*)d_in[6];

    constexpr int Bb = 2, S = 2048, DM = 2048, HKV = 8, DK = 64;
    const int M   = Bb * S;          // 4096
    const int Nkv = HKV * DK;        // 512

    size_t off = 0;
    auto alloc = [&](size_t bytes) -> char* {
        char* p = (char*)d_ws + off;
        off += (bytes + 255) & ~(size_t)255;
        return p;
    };
    u16* Wqb = (u16*)alloc((size_t)DM * DM * 2);
    u16* Wkb = (u16*)alloc((size_t)Nkv * DM * 2);
    u16* Wvb = (u16*)alloc((size_t)Nkv * DM * 2);
    u16* Wob = (u16*)alloc((size_t)DM * DM * 2);
    u16* Qb  = (u16*)alloc((size_t)M * DM * 2);
    u16* Kb  = (u16*)alloc((size_t)M * Nkv * 2);
    u16* Vt  = (u16*)alloc((size_t)M * Nkv * 2);
    u16* Ob  = (u16*)alloc((size_t)M * DM * 2);

    // weights -> bf16
    cvt_f32_bf16<<<512, 256, 0, stream>>>(W_q, Wqb, DM * DM);
    cvt_f32_bf16<<<512, 256, 0, stream>>>(W_k, Wkb, Nkv * DM);
    cvt_f32_bf16<<<512, 256, 0, stream>>>(W_v, Wvb, Nkv * DM);
    cvt_f32_bf16<<<512, 256, 0, stream>>>(W_o, Wob, DM * DM);

    // projections (A fp32 converted on load)
    dim3 blk(256);
    gemm_wmma_bf16<<<dim3(DM / BN, M / BM), blk, 0, stream>>>(
        query, nullptr, Wqb, Qb, M, DM, DM, /*mode=*/0, S);
    gemm_wmma_bf16<<<dim3(Nkv / BN, M / BM), blk, 0, stream>>>(
        key, nullptr, Wkb, Kb, M, Nkv, DM, /*mode=*/0, S);
    gemm_wmma_bf16<<<dim3(Nkv / BN, M / BM), blk, 0, stream>>>(
        value, nullptr, Wvb, Vt, M, Nkv, DM, /*mode=*/2, S);

    // grouped-query flash attention
    gqa_flash_attn<<<dim3(S / 64, Bb * 32), dim3(128), 0, stream>>>(Qb, Kb, Vt, Ob);

    // output projection -> fp32 d_out
    gemm_wmma_bf16<<<dim3(DM / BN, M / BM), blk, 0, stream>>>(
        nullptr, Ob, Wob, d_out, M, DM, DM, /*mode=*/1, S);
}
